// Inference_44143673868297
// MI455X (gfx1250) — matrix operand order repaired and checked by reading, not compiled, not run on hardware
//
#include <hip/hip_runtime.h>
#include <hip/hip_bf16.h>

typedef __attribute__((ext_vector_type(16))) _Float16 v16h;
typedef __attribute__((ext_vector_type(8)))  float    v8f;

// ---------------------------------------------------------------------------
// Problem constants (from reference)
// DIM=128, NUM_HEADS=4, HEAD_DIM=32, WS=8, M=64, N_CAND=4, L=256, NW=B_=128
// ---------------------------------------------------------------------------
static constexpr int   kB    = 128;   // windows (batch)
static constexpr int   kL    = 256;   // tokens per window
static constexpr int   kH    = 4;     // heads
static constexpr int   kHD   = 32;    // head dim
static constexpr int   kDIM  = 128;
static constexpr int   kTOK  = 64;    // spatial tokens per window (M)
static constexpr float kSCALE = 0.17677669529663687f; // 32^-0.5

// Wave id as a compiler-provable scalar (SGPR): wave-uniform by construction.
__device__ inline int wave_id() {
  return __builtin_amdgcn_readfirstlane((int)(threadIdx.x >> 5));
}

// ---------------------------------------------------------------------------
// WMMA helpers (gfx1250, wave32). D = A(16x32 f16) * B(32x16 f16) + C(f32)
// Layouts per CDNA5 ISA 7.12.2:
//   A 16x32 f16 : lane l -> row m=l&15; vgpr v: K = (v<4 ? (l>>4)*8+2v
//                                                      : 16+(l>>4)*8+2(v-4)), K+1
//   B 32x16 f16 : lane l -> col n=l&15; same K order as A (elem s: K=(l>>4)*8+s, s>=8: +8)
//   C/D 16x16 f32: vgpr g, lane l -> m = g + 8*(l>>4), n = l&15
// ---------------------------------------------------------------------------
__device__ inline v8f vzero8() {
  v8f z;
#pragma unroll
  for (int i = 0; i < 8; ++i) z[i] = 0.0f;
  return z;
}

template <typename RowPtrF>
__device__ inline v16h ld_frag_a_f16(RowPtrF rowptr) {
  const int l  = threadIdx.x & 31;
  const int m  = l & 15;
  const int hs = l >> 4;
  const _Float16* r = rowptr(m);
  v16h a;
#pragma unroll
  for (int v = 0; v < 8; ++v) {
    const int kb = (v < 4) ? (hs * 8 + 2 * v) : (16 + hs * 8 + 2 * (v - 4));
    a[2 * v]     = r[kb];
    a[2 * v + 1] = r[kb + 1];
  }
  return a;
}

template <typename RowPtrF>
__device__ inline v16h ld_frag_a_f32(RowPtrF rowptr) {  // f32 source (LDS scores)
  const int l  = threadIdx.x & 31;
  const int m  = l & 15;
  const int hs = l >> 4;
  const float* r = rowptr(m);
  v16h a;
#pragma unroll
  for (int v = 0; v < 8; ++v) {
    const int kb = (v < 4) ? (hs * 8 + 2 * v) : (16 + hs * 8 + 2 * (v - 4));
    a[2 * v]     = (_Float16)r[kb];
    a[2 * v + 1] = (_Float16)r[kb + 1];
  }
  return a;
}

template <typename ColPtrF>
__device__ inline v16h ld_frag_b_f16(ColPtrF colptr) {
  const int l  = threadIdx.x & 31;
  const int n  = l & 15;
  const int ks = (l >> 4) * 8;
  const _Float16* r = colptr(n) + ks;
  v16h b;
#pragma unroll
  for (int v = 0; v < 16; ++v) b[v] = r[v < 8 ? v : v + 8];
  return b;
}

__device__ inline v8f wmma16(v16h a, v16h b, v8f c) {
  return __builtin_amdgcn_wmma_f32_16x16x32_f16(
      /*neg_a=*/false, a, /*neg_b=*/false, b,
      /*c_mod=*/(short)0, c, /*reuse_a=*/false, /*reuse_b=*/false);
}

// ---------------------------------------------------------------------------
// Prep kernels
// ---------------------------------------------------------------------------
__global__ void k_cvt(const float* __restrict__ src, _Float16* __restrict__ dst, int n) {
  int i = blockIdx.x * blockDim.x + threadIdx.x;
  if (i < n) dst[i] = (_Float16)src[i];
}

// rpe_table row layout (384): [h][3*hd] -> q=row[h*96+c], k=row[h*96+32+c], v=row[h*96+64+c]
__global__ void k_prep_rpe(const float* __restrict__ rpe_table,
                           const int* __restrict__ rel_idx,
                           _Float16* __restrict__ q_rpe,   // [jm][h][im][c]  (B for kr)
                           _Float16* __restrict__ k_rpe,   // [im][h][jm][c]  (B for qr)
                           _Float16* __restrict__ v_rpe) { // [im][h][c][jm]  (B for out2)
  int t = blockIdx.x * blockDim.x + threadIdx.x;
  if (t >= kTOK * kTOK * kH * kHD) return;
  const int c  = t & 31;
  const int h  = (t >> 5) & 3;
  const int jm = (t >> 7) & 63;
  const int im = t >> 13;
  const int id = rel_idx[im * kTOK + jm];
  const float* row = rpe_table + (size_t)id * 384 + h * 96 + c;
  q_rpe[((size_t)(jm * kH + h) * kTOK + im) * kHD + c] = (_Float16)(row[0] * kSCALE);
  k_rpe[((size_t)(im * kH + h) * kTOK + jm) * kHD + c] = (_Float16)row[32];
  v_rpe[((size_t)(im * kH + h) * kHD + c) * kTOK + jm] = (_Float16)row[64];
}

// ---------------------------------------------------------------------------
// Kernel 1: QKV = x @ qkv_w^T + qkv_b   (rows = 32768, N = 384, K = 128)
// Outputs: q16[b][h][i][c] (*scale), k16[b][h][j][c], vT16[b][h][c][j]
// A fragments hoisted; nt/part are SGPR-uniform -> scalar store branches.
// ---------------------------------------------------------------------------
__global__ void __launch_bounds__(256) k_qkv(const _Float16* __restrict__ x16,
                                             const _Float16* __restrict__ w16,  // [384][128]
                                             const float* __restrict__ bias,    // [384]
                                             _Float16* __restrict__ q16,
                                             _Float16* __restrict__ k16,
                                             _Float16* __restrict__ vT16) {
  const int rt   = blockIdx.x;          // 2048 row tiles of 16
  const int wave = wave_id();
  const int lane = threadIdx.x & 31;
  const int r0   = rt * 16;

  v16h af[4];
#pragma unroll
  for (int kk = 0; kk < 4; ++kk)
    af[kk] = ld_frag_a_f16([&](int m) { return x16 + (size_t)(r0 + m) * kDIM + kk * 32; });

#pragma unroll
  for (int it = 0; it < 3; ++it) {      // nt = wave, wave+8, wave+16 -> part = it
    const int nt = wave + it * 8;
    const int n0 = nt * 16;
    v16h bf[4];
#pragma unroll
    for (int kk = 0; kk < 4; ++kk)
      bf[kk] = ld_frag_b_f16([&](int n) { return w16 + (size_t)(n0 + n) * kDIM + kk * 32; });
    v8f acc = vzero8();
#pragma unroll
    for (int kk = 0; kk < 4; ++kk) acc = wmma16(af[kk], bf[kk], acc);

    const int nl   = n0 + (lane & 15);
    const int h    = (nl >> 5) & 3;
    const int c    = nl & 31;
    const float bv = bias[nl];
    if (it == 0) {          // q (scaled)
#pragma unroll
      for (int g = 0; g < 8; ++g) {
        const int r = r0 + g + 8 * (lane >> 4);
        const int b_ = r >> 8, i = r & 255;
        q16[((size_t)(b_ * kH + h) * kL + i) * kHD + c] = (_Float16)((acc[g] + bv) * kSCALE);
      }
    } else if (it == 1) {   // k
#pragma unroll
      for (int g = 0; g < 8; ++g) {
        const int r = r0 + g + 8 * (lane >> 4);
        const int b_ = r >> 8, i = r & 255;
        k16[((size_t)(b_ * kH + h) * kL + i) * kHD + c] = (_Float16)(acc[g] + bv);
      }
    } else {                // v, stored transposed [b][h][c][j]
#pragma unroll
      for (int g = 0; g < 8; ++g) {
        const int r = r0 + g + 8 * (lane >> 4);
        const int b_ = r >> 8, i = r & 255;
        vT16[((size_t)(b_ * kH + h) * kHD + c) * kL + i] = (_Float16)(acc[g] + bv);
      }
    }
  }
}

// ---------------------------------------------------------------------------
// Kernel 2: qr_m[b][h][i][jm] = q[b,h,i,:] . k_rpe[i/4, jm, h, :]
// Batched over b (16 windows share one B matrix).  grid = (h,i) = 1024 WGs.
// ---------------------------------------------------------------------------
__global__ void __launch_bounds__(256) k_qr(const _Float16* __restrict__ q16,
                                            const _Float16* __restrict__ k_rpe, // [im][h][jm][c]
                                            _Float16* __restrict__ qr_m) {      // [b][h][i][jm]
  const int h    = blockIdx.x & 3;
  const int i    = blockIdx.x >> 2;     // 0..255
  const int im   = i >> 2;
  const int wave = wave_id();
  const int lane = threadIdx.x & 31;
  const int bt   = wave;                // b-tile = wave (8 tiles of 16 windows)
  v16h a = ld_frag_a_f16([&](int m) {
    return q16 + (((size_t)(bt * 16 + m) * kH + h) * kL + i) * kHD;
  });
  v16h bf[4];
#pragma unroll
  for (int jt = 0; jt < 4; ++jt)
    bf[jt] = ld_frag_b_f16([&](int n) {
      return k_rpe + ((size_t)(im * kH + h) * kTOK + jt * 16 + n) * kHD;
    });
#pragma unroll
  for (int jt = 0; jt < 4; ++jt) {
    v8f acc = vzero8();
    acc = wmma16(a, bf[jt], acc);
    const int jm = jt * 16 + (lane & 15);
#pragma unroll
    for (int g = 0; g < 8; ++g) {
      const int bb = bt * 16 + g + 8 * (lane >> 4);
      qr_m[(((size_t)(bb * kH + h) * kL + i) * kTOK) + jm] = (_Float16)acc[g];
    }
  }
}

// ---------------------------------------------------------------------------
// Kernel 3: kr_m[b][h][im][j] = k[b,h,j,:] . q_rpe[im, j/4, h, :]  (q_rpe scaled)
// grid = (h,jm) = 256 WGs; rows = (b, j%4) = 512, N = im (64).
// ---------------------------------------------------------------------------
__global__ void __launch_bounds__(256) k_kr(const _Float16* __restrict__ k16,
                                            const _Float16* __restrict__ q_rpe, // [jm][h][im][c]
                                            _Float16* __restrict__ kr_m) {      // [b][h][im][j]
  const int h    = blockIdx.x & 3;
  const int jm   = blockIdx.x >> 2;     // 0..63
  const int wave = wave_id();
  const int lane = threadIdx.x & 31;
  // B fragments (4 im tiles) shared across all row tiles of this WG: preload.
  v16h bf[4];
#pragma unroll
  for (int it = 0; it < 4; ++it)
    bf[it] = ld_frag_b_f16([&](int n) {
      return q_rpe + ((size_t)(jm * kH + h) * kTOK + it * 16 + n) * kHD;
    });
#pragma unroll
  for (int u = 0; u < 4; ++u) {
    const int rt = wave * 4 + u;        // 32 row tiles total
    v16h a = ld_frag_a_f16([&](int m) {
      const int r = rt * 16 + m;
      const int bb = r >> 2, j4 = r & 3;
      return k16 + (((size_t)(bb * kH + h) * kL) + jm * 4 + j4) * kHD;
    });
#pragma unroll
    for (int it = 0; it < 4; ++it) {
      v8f acc = vzero8();
      acc = wmma16(a, bf[it], acc);
      const int imv = it * 16 + (lane & 15);
#pragma unroll
      for (int g = 0; g < 8; ++g) {
        const int r = rt * 16 + g + 8 * (lane >> 4);
        const int bb = r >> 2, j4 = r & 3;
        kr_m[(((size_t)(bb * kH + h) * kTOK + imv) * kL) + jm * 4 + j4] = (_Float16)acc[g];
      }
    }
  }
}

// ---------------------------------------------------------------------------
// Kernel 4: fused attention per (b, h, 32-row i-block).
//   scores = q@k^T + qr + kr + mask -> softmax(f32 in LDS)
//   out1 = attn @ v ; attn_sum[h][im][b][i4][jm] = candidate-group row sums
// ---------------------------------------------------------------------------
__global__ void __launch_bounds__(256) k_attn(const _Float16* __restrict__ q16,
                                              const _Float16* __restrict__ k16,
                                              const _Float16* __restrict__ vT16,
                                              const _Float16* __restrict__ qr_m,
                                              const _Float16* __restrict__ kr_m,
                                              const float* __restrict__ mask,   // [128][256][256]
                                              float* __restrict__ out1,         // [b][h][i][c]
                                              _Float16* __restrict__ attn_sum) {
  __shared__ float sc[32 * 256];
  __shared__ float red[32 * 8];
  __shared__ float rstat[32];

  const int bh  = blockIdx.x >> 3;
  const int ibl = blockIdx.x & 7;
  const int b   = bh >> 2, h = bh & 3;
  const int wave = wave_id();
  const int lane = threadIdx.x & 31;
  const size_t qbase = ((size_t)(b * kH + h)) * kL * kHD;
  const int i0base = ibl * 32;

  // ---- Phase 1: score tiles (qk WMMA) + qr + kr + mask -> LDS (f32) ----
  {
    const int s  = wave >> 2;           // i strip (2 x 16 rows)
    const int jq = wave & 3;            // j quarter (4 jtiles each)
    const int i0 = i0base + s * 16;
    v16h a = ld_frag_a_f16([&](int m) { return q16 + qbase + (size_t)(i0 + m) * kHD; });
    v16h bfr[4];
#pragma unroll
    for (int u = 0; u < 4; ++u)
      bfr[u] = ld_frag_b_f16([&](int n) {
        return k16 + qbase + (size_t)((jq * 4 + u) * 16 + n) * kHD;
      });
#pragma unroll
    for (int u = 0; u < 4; ++u) {
      const int j0 = (jq * 4 + u) * 16;
      v8f acc = vzero8();
      acc = wmma16(a, bfr[u], acc);
      const int j = j0 + (lane & 15);
#pragma unroll
      for (int g = 0; g < 8; ++g) {
        const int li = s * 16 + g + 8 * (lane >> 4);
        const int gi = i0base + li;
        float v = acc[g];
        v += (float)qr_m[(((size_t)(b * kH + h) * kL + gi) * kTOK) + (j >> 2)];
        v += (float)kr_m[(((size_t)(b * kH + h) * kTOK + (gi >> 2)) * kL) + j];
        v += mask[((size_t)b * kL + gi) * kL + j];
        sc[li * 256 + j] = v;
      }
    }
  }
  __syncthreads();

  // ---- Phase 2: row softmax (8 threads per row, seg of 32) ----
  const int li  = threadIdx.x >> 3;
  const int seg = threadIdx.x & 7;
  {
    float mx = -3.4e38f;
    for (int j = seg * 32; j < seg * 32 + 32; ++j) mx = fmaxf(mx, sc[li * 256 + j]);
    red[li * 8 + seg] = mx;
  }
  __syncthreads();
  if (seg == 0) {
    float mx = red[li * 8];
#pragma unroll
    for (int s2 = 1; s2 < 8; ++s2) mx = fmaxf(mx, red[li * 8 + s2]);
    rstat[li] = mx;
  }
  __syncthreads();
  {
    const float mx = rstat[li];
    float sum = 0.f;
    for (int j = seg * 32; j < seg * 32 + 32; ++j) {
      const float e = __expf(sc[li * 256 + j] - mx);
      sc[li * 256 + j] = e;
      sum += e;
    }
    red[li * 8 + seg] = sum;
  }
  __syncthreads();
  if (seg == 0) {
    float s2 = 0.f;
#pragma unroll
    for (int t = 0; t < 8; ++t) s2 += red[li * 8 + t];
    rstat[li] = 1.0f / s2;
  }
  __syncthreads();
  {
    const float inv = rstat[li];
    const int gi = i0base + li;
    const int im = gi >> 2, i4 = gi & 3;
    for (int jm = seg * 8; jm < seg * 8 + 8; ++jm) {
      float s0 = sc[li * 256 + jm * 4 + 0] * inv;
      float s1 = sc[li * 256 + jm * 4 + 1] * inv;
      float s2 = sc[li * 256 + jm * 4 + 2] * inv;
      float s3 = sc[li * 256 + jm * 4 + 3] * inv;
      sc[li * 256 + jm * 4 + 0] = s0;
      sc[li * 256 + jm * 4 + 1] = s1;
      sc[li * 256 + jm * 4 + 2] = s2;
      sc[li * 256 + jm * 4 + 3] = s3;
      attn_sum[((((size_t)(h * kTOK + im)) * kB + b) * 4 + i4) * kTOK + jm] =
          (_Float16)(s0 + s1 + s2 + s3);
    }
  }
  __syncthreads();

  // ---- Phase 3: out1 = attn @ v (waves 0..3) ----
  if (wave < 4) {
    const int s  = wave >> 1;
    const int ct = wave & 1;
    v8f acc = vzero8();
    for (int kk = 0; kk < 8; ++kk) {
      const int k0 = kk * 32;
      v16h a  = ld_frag_a_f32([&](int m) { return &sc[(s * 16 + m) * 256 + k0]; });
      v16h bf = ld_frag_b_f16([&](int n) {
        return vT16 + ((size_t)(b * kH + h) * kHD + ct * 16 + n) * kL + k0;
      });
      acc = wmma16(a, bf, acc);
    }
    const int c = ct * 16 + (lane & 15);
#pragma unroll
    for (int g = 0; g < 8; ++g) {
      const int gi = i0base + s * 16 + g + 8 * (lane >> 4);
      out1[((size_t)(b * kH + h) * kL + gi) * kHD + c] = acc[g];
    }
  }
}

// ---------------------------------------------------------------------------
// Kernel 5: out2[b,h,i,c] = sum_jm attn_sum * v_rpe ; merge with out1 ->
//           out_heads[b*256+i][h*32+c] (f16).  grid = (h,im) = 256 WGs.
// ---------------------------------------------------------------------------
__global__ void __launch_bounds__(256) k_out2(const _Float16* __restrict__ attn_sum,
                                              const _Float16* __restrict__ v_rpe, // [im][h][c][jm]
                                              const float* __restrict__ out1,
                                              _Float16* __restrict__ out_heads) {
  const int h  = blockIdx.x & 3;
  const int im = blockIdx.x >> 2;   // 0..63
  const int wave = wave_id();
  const int lane = threadIdx.x & 31;
  const _Float16* abase = attn_sum + ((size_t)(h * kTOK + im)) * kB * 4 * kTOK;
  // Preload both c-tile B fragments (K = 64 -> 2 frags each), shared by all rows.
  v16h bf[2][2];
#pragma unroll
  for (int ct = 0; ct < 2; ++ct)
#pragma unroll
    for (int kk = 0; kk < 2; ++kk)
      bf[ct][kk] = ld_frag_b_f16([&](int n) {
        return v_rpe + ((size_t)(im * kH + h) * kHD + ct * 16 + n) * kTOK + kk * 32;
      });
#pragma unroll
  for (int u = 0; u < 4; ++u) {
    const int rt = wave * 4 + u;      // 32 row tiles total
    v16h a0 = ld_frag_a_f16([&](int m) { return abase + (size_t)(rt * 16 + m) * kTOK; });
    v16h a1 = ld_frag_a_f16([&](int m) { return abase + (size_t)(rt * 16 + m) * kTOK + 32; });
#pragma unroll
    for (int ct = 0; ct < 2; ++ct) {
      v8f acc = vzero8();
      acc = wmma16(a0, bf[ct][0], acc);
      acc = wmma16(a1, bf[ct][1], acc);
      const int c = ct * 16 + (lane & 15);
#pragma unroll
      for (int g = 0; g < 8; ++g) {
        const int r = rt * 16 + g + 8 * (lane >> 4);
        const int bb = r >> 2, i4 = r & 3, i = im * 4 + i4;
        const float v = acc[g] + out1[((size_t)(bb * kH + h) * kL + i) * kHD + c];
        out_heads[((size_t)bb * kL + i) * kDIM + h * kHD + c] = (_Float16)v;
      }
    }
  }
}

// ---------------------------------------------------------------------------
// Kernel 6: final projection  out = out_heads @ proj_w^T + proj_b  (f32 out)
// ---------------------------------------------------------------------------
__global__ void __launch_bounds__(256) k_proj(const _Float16* __restrict__ oh,   // [32768][128]
                                              const _Float16* __restrict__ w16,  // [128][128]
                                              const float* __restrict__ bias,    // [128]
                                              float* __restrict__ out) {
  const int rt   = blockIdx.x;      // 2048
  const int wave = wave_id();
  const int lane = threadIdx.x & 31;
  const int n0 = wave * 16;         // 8 waves cover 128 output cols
  v16h af[4], bf[4];
#pragma unroll
  for (int kk = 0; kk < 4; ++kk) {
    af[kk] = ld_frag_a_f16([&](int m) { return oh + (size_t)(rt * 16 + m) * kDIM + kk * 32; });
    bf[kk] = ld_frag_b_f16([&](int n) { return w16 + (size_t)(n0 + n) * kDIM + kk * 32; });
  }
  v8f acc = vzero8();
#pragma unroll
  for (int kk = 0; kk < 4; ++kk) acc = wmma16(af[kk], bf[kk], acc);
  const int n = n0 + (lane & 15);
  const float bv = bias[n];
#pragma unroll
  for (int g = 0; g < 8; ++g) {
    const int r = rt * 16 + g + 8 * (lane >> 4);
    out[(size_t)r * kDIM + n] = acc[g] + bv;
  }
}

// ---------------------------------------------------------------------------
// Host launch
// ---------------------------------------------------------------------------
extern "C" void kernel_launch(void* const* d_in, const int* in_sizes, int n_in,
                              void* d_out, int out_size, void* d_ws, size_t ws_size,
                              hipStream_t stream) {
  const float* x       = (const float*)d_in[0];  // [128,256,128]
  const float* qkv_w   = (const float*)d_in[1];  // [384,128]
  const float* qkv_b   = (const float*)d_in[2];  // [384]
  const float* rpe_tab = (const float*)d_in[3];  // [225,384]
  const float* proj_w  = (const float*)d_in[4];  // [128,128]
  const float* proj_b  = (const float*)d_in[5];  // [128]
  const float* mask    = (const float*)d_in[6];  // [128,256,256]
  const int*   rel_idx = (const int*)d_in[7];    // [64,64]
  float* out = (float*)d_out;

  char* wsb = (char*)d_ws;
  size_t off = 0;
  auto alloc = [&](size_t bytes) -> void* {
    void* p = wsb + off;
    off += (bytes + 255) & ~(size_t)255;
    return p;
  };

  const size_t nTok = (size_t)kB * kL;                 // 32768
  _Float16* x16   = (_Float16*)alloc(nTok * kDIM * 2);
  _Float16* w16   = (_Float16*)alloc((size_t)384 * 128 * 2);
  _Float16* pw16  = (_Float16*)alloc((size_t)128 * 128 * 2);
  _Float16* q16   = (_Float16*)alloc(nTok * kDIM * 2);
  _Float16* k16   = (_Float16*)alloc(nTok * kDIM * 2);
  _Float16* vT16  = (_Float16*)alloc(nTok * kDIM * 2);
  _Float16* q_rpe = (_Float16*)alloc((size_t)64 * 4 * 64 * 32 * 2);
  _Float16* k_rpe = (_Float16*)alloc((size_t)64 * 4 * 64 * 32 * 2);
  _Float16* v_rpe = (_Float16*)alloc((size_t)64 * 4 * 64 * 32 * 2);
  _Float16* qr_m  = (_Float16*)alloc(nTok * 4 * 64 * 2);
  _Float16* kr_m  = (_Float16*)alloc(nTok * 4 * 64 * 2);
  float*    out1  = (float*)alloc(nTok * kDIM * 4);
  _Float16* asum  = (_Float16*)alloc(nTok * 4 * 64 * 2);
  _Float16* oh    = (_Float16*)alloc(nTok * kDIM * 2);
  (void)ws_size; (void)in_sizes; (void)n_in; (void)out_size;

  k_cvt<<<(int)((nTok * kDIM + 255) / 256), 256, 0, stream>>>(x, x16, (int)(nTok * kDIM));
  k_cvt<<<(384 * 128 + 255) / 256, 256, 0, stream>>>(qkv_w, w16, 384 * 128);
  k_cvt<<<(128 * 128 + 255) / 256, 256, 0, stream>>>(proj_w, pw16, 128 * 128);
  k_prep_rpe<<<(64 * 64 * 4 * 32 + 255) / 256, 256, 0, stream>>>(rpe_tab, rel_idx,
                                                                 q_rpe, k_rpe, v_rpe);
  k_qkv<<<2048, 256, 0, stream>>>(x16, w16, qkv_b, q16, k16, vT16);
  k_qr<<<1024, 256, 0, stream>>>(q16, k_rpe, qr_m);
  k_kr<<<256, 256, 0, stream>>>(k16, q_rpe, kr_m);
  k_attn<<<4096, 256, 0, stream>>>(q16, k16, vT16, qr_m, kr_m, mask, out1, asum);
  k_out2<<<256, 256, 0, stream>>>(asum, v_rpe, out1, oh);
  k_proj<<<2048, 256, 0, stream>>>(oh, pw16, proj_b, out);
}
